// MultiheadAttention_60859686584835
// MI455X (gfx1250) — matrix operand order repaired and checked by reading, not compiled, not run on hardware
//
#include <hip/hip_runtime.h>
#include <hip/hip_bf16.h>

typedef __attribute__((ext_vector_type(16))) _Float16 v16h;
typedef __attribute__((ext_vector_type(8)))  _Float16 v8h;
typedef __attribute__((ext_vector_type(8)))  float    v8f;

static __device__ __forceinline__ v16h cat16(v8h a, v8h b) {
    return __builtin_shufflevector(a, b, 0,1,2,3,4,5,6,7,8,9,10,11,12,13,14,15);
}

static __device__ __forceinline__ v8f wmma_f16(v16h a, v16h b, v8f c) {
    // v_wmma_f32_16x16x32_f16
    return __builtin_amdgcn_wmma_f32_16x16x32_f16(false, a, false, b, (short)0, c, false, false);
}

// ---------------------------------------------------------------------------
// f32 -> f16 conversion pass (query/key/value), 8 elements per thread.
// ---------------------------------------------------------------------------
__global__ __launch_bounds__(256) void cvt_kernel(const float* __restrict__ src,
                                                  _Float16* __restrict__ dst) {
    const int i = (blockIdx.x * 256 + threadIdx.x) * 8;
    #pragma unroll
    for (int k = 0; k < 8; ++k) dst[i + k] = (_Float16)src[i + k];
}

// ---------------------------------------------------------------------------
// Bias-count table: one int per (b,i,j)
//   bits[ 0: 9] : # parentchild entries (b,i,j)          -> emb[0]
//   bits[10:19] : # parentchild entries (b,j,i)          -> emb[1]  (pre-transposed)
//   bits[20:29] : # sibling entries (b,i,j)+(b,j,i)      -> emb[2]
// ---------------------------------------------------------------------------
__global__ __launch_bounds__(256) void count_kernel(const int* __restrict__ pc,
                                                    const int* __restrict__ sb,
                                                    int* __restrict__ cnt) {
    int idx = blockIdx.x * 256 + threadIdx.x;
    if (idx < 8192) {
        int b = pc[idx*3], i = pc[idx*3+1], j = pc[idx*3+2];
        atomicAdd(&cnt[(b<<20) + (i<<10) + j], 1);
        atomicAdd(&cnt[(b<<20) + (j<<10) + i], 1 << 10);
    } else if (idx < 16384) {
        int t = idx - 8192;
        int b = sb[t*3], i = sb[t*3+1], j = sb[t*3+2];
        atomicAdd(&cnt[(b<<20) + (i<<10) + j], 1 << 20);
        atomicAdd(&cnt[(b<<20) + (j<<10) + i], 1 << 20);
    }
}

// ---------------------------------------------------------------------------
// WMMA GEMM: C(M x 512) = A(M x 512) * W(512 x 512) + bias, A already f16.
// A tile staged into LDS with GLOBAL_LOAD_ASYNC_TO_LDS_B128 (ASYNCcnt path).
//   KIND 0: out f16 row-major, scaled by 0.125 (Q, folds 1/sqrt(DH))
//   KIND 1: out f16 row-major                   (K)
//   KIND 2: out f16 transposed (b,h,dh,s)       (V^T)
//   KIND 3: out f32 row-major                   (final X @ Wo + bo)
// Block: 256 threads = 8 waves, tile 128(M) x 64(N), k-step 32.
// ---------------------------------------------------------------------------
template <int KIND>
__global__ __launch_bounds__(256) void gemm_kernel(const _Float16* __restrict__ A,
                                                   const float* __restrict__ W,
                                                   const float* __restrict__ bias,
                                                   void* __restrict__ Out) {
    __shared__ _Float16 lA[128 * 32];   // 8 KB, row-major [m][k]
    __shared__ _Float16 lB[64 * 32];    // 4 KB, transposed [n][k]

    const int tid  = threadIdx.x;
    const int lane = tid & 31;
    const int wid  = tid >> 5;                  // 0..7
    const int wm   = wid >> 1;                  // 0..3  (row sub-block)
    const int wn   = wid & 1;                   // 0..1  (col sub-block)
    const int m0   = blockIdx.x * 128;
    const int n0   = blockIdx.y * 64;

    v8f C[2][2] = {{{}, {}}, {{}, {}}};

    const int nl = lane & 15;
    const int hi = lane >> 4;

    for (int k0 = 0; k0 < 512; k0 += 32) {
        __syncthreads();
        // ---- stage A tile (128 x 32 f16) via async copy to LDS ----
        {
            const int r  = tid >> 1;
            const int c0 = (tid & 1) * 16;
            const _Float16* src = A + (size_t)(m0 + r) * 512 + k0 + c0;
            // LDS byte offset = low 32 bits of the generic shared-aperture address
            unsigned ldsoff = (unsigned)(unsigned long long)(const void*)&lA[r * 32 + c0];
            unsigned long long ga = (unsigned long long)src;
            asm volatile(
                "global_load_async_to_lds_b128 %0, %1, off\n\t"
                "global_load_async_to_lds_b128 %0, %1, off offset:16"
                :: "v"(ldsoff), "v"(ga) : "memory");
        }
        // ---- stage W tile (32 x 64 f32 -> f16) transposed into lB[n][k] ----
        {
            const int kk = tid & 31;
            const int nb = (tid >> 5) * 8;
            const float* src = W + (size_t)(k0 + kk) * 512 + n0 + nb;
            __builtin_prefetch(src + 32 * 512, 0, 0);   // global_prefetch_b8 next k-tile
            #pragma unroll
            for (int i = 0; i < 8; ++i)
                lB[(nb + i) * 32 + kk] = (_Float16)src[i];
        }
        asm volatile("s_wait_asynccnt 0" ::: "memory");
        __syncthreads();

        // ---- fragments + 4 WMMAs ----
        v16h Afr[2], Bfr[2];
        const int koffA = (lane < 16) ? 0 : 8;
        const int koffB = (lane < 16) ? 0 : 8;
        #pragma unroll
        for (int s = 0; s < 2; ++s) {
            const int row = wm * 32 + s * 16 + nl;
            v8h a0 = *(const v8h*)&lA[row * 32 + koffA];
            v8h a1 = *(const v8h*)&lA[row * 32 + koffA + 16];
            Afr[s] = cat16(a0, a1);
        }
        #pragma unroll
        for (int t = 0; t < 2; ++t) {
            const int col = wn * 32 + t * 16 + nl;
            v8h b0 = *(const v8h*)&lB[col * 32 + koffB];
            v8h b1 = *(const v8h*)&lB[col * 32 + koffB + 16];
            Bfr[t] = cat16(b0, b1);
        }
        #pragma unroll
        for (int s = 0; s < 2; ++s)
            #pragma unroll
            for (int t = 0; t < 2; ++t)
                C[s][t] = wmma_f16(Afr[s], Bfr[t], C[s][t]);
    }

    // ---- store ----
    #pragma unroll
    for (int s = 0; s < 2; ++s) {
        #pragma unroll
        for (int t = 0; t < 2; ++t) {
            #pragma unroll
            for (int r = 0; r < 8; ++r) {
                const int gm = m0 + wm * 32 + s * 16 + r + 8 * hi;
                const int gn = n0 + wn * 32 + t * 16 + nl;
                float v = C[s][t][r] + bias[gn];
                if constexpr (KIND == 0) {
                    ((_Float16*)Out)[(size_t)gm * 512 + gn] = (_Float16)(v * 0.125f);
                } else if constexpr (KIND == 1) {
                    ((_Float16*)Out)[(size_t)gm * 512 + gn] = (_Float16)v;
                } else if constexpr (KIND == 2) {
                    // V^T layout: vt[(b*512 + n) * 1024 + s], b = gm>>10, s = gm&1023
                    ((_Float16*)Out)[((size_t)((gm >> 10) * 512 + gn) << 10) + (gm & 1023)] = (_Float16)v;
                } else {
                    ((float*)Out)[(size_t)gm * 512 + gn] = v;
                }
            }
        }
    }
}

// ---------------------------------------------------------------------------
// Flash attention with relational bias. One wave per (b, h, 16-row i-tile).
// Q pre-scaled by 1/sqrt(DH). j processed in 32-wide tiles.
// ---------------------------------------------------------------------------
__global__ __launch_bounds__(256) void attn_kernel(const _Float16* __restrict__ Q,
                                                   const _Float16* __restrict__ K,
                                                   const _Float16* __restrict__ Vt,
                                                   const int* __restrict__ cnt,
                                                   const float* __restrict__ emb,
                                                   _Float16* __restrict__ X) {
    __shared__ _Float16 lP[8 * 512];   // 1 KB P-staging slot per wave

    const int tid  = threadIdx.x;
    const int lane = tid & 31;
    const int wid  = tid >> 5;
    const int w    = blockIdx.x * 8 + wid;      // 0..4095
    const int it   = w & 63;
    const int h    = (w >> 6) & 7;
    const int b    = w >> 9;

    const int nl = lane & 15;
    const int hi = lane >> 4;

    const float e0 = emb[h], e1 = emb[8 + h], e2 = emb[16 + h];

    // ---- Q fragments (16 x 64 = two 16x32 A-frags), loaded once ----
    v16h Qfr[2];
    {
        const int koff = (lane < 16) ? 0 : 8;
        const _Float16* qp = Q + ((size_t)(b * 1024 + it * 16 + nl)) * 512 + h * 64;
        #pragma unroll
        for (int s = 0; s < 2; ++s) {
            v8h a0 = *(const v8h*)(qp + s * 32 + koff);
            v8h a1 = *(const v8h*)(qp + s * 32 + koff + 16);
            Qfr[s] = cat16(a0, a1);
        }
    }

    v8f O[4] = {{}, {}, {}, {}};
    float mrow[8], lrow[8];
    #pragma unroll
    for (int r = 0; r < 8; ++r) { mrow[r] = -3.0e38f; lrow[r] = 0.0f; }

    const _Float16* pslot = &lP[wid * 512];

    for (int jt = 0; jt < 32; ++jt) {
        const int j0 = jt * 32;

        // ---- scores S = Q K^T (two 16x16 f32 frags over 32 j-columns) ----
        v8f S[2] = {{}, {}};
        {
            const int kb = (lane < 16) ? 0 : 8;
            #pragma unroll
            for (int t = 0; t < 2; ++t) {
                const _Float16* kp = K + ((size_t)(b * 1024 + j0 + t * 16 + nl)) * 512 + h * 64;
                #pragma unroll
                for (int s = 0; s < 2; ++s) {
                    v8h b0 = *(const v8h*)(kp + s * 32 + kb);
                    v8h b1 = *(const v8h*)(kp + s * 32 + kb + 16);
                    S[t] = wmma_f16(Qfr[s], cat16(b0, b1), S[t]);
                }
            }
        }

        // ---- relational bias from packed count table ----
        float sv[16];
        #pragma unroll
        for (int t = 0; t < 2; ++t) {
            #pragma unroll
            for (int r = 0; r < 8; ++r) {
                const int gi = it * 16 + r + 8 * hi;
                const int gj = j0 + t * 16 + nl;
                const int c  = cnt[((size_t)b << 20) + ((size_t)gi << 10) + gj];
                sv[t * 8 + r] = S[t][r]
                              + (float)(c & 1023)          * e0
                              + (float)((c >> 10) & 1023)  * e1
                              + (float)((c >> 20) & 1023)  * e2;
            }
        }

        // ---- online softmax (row reduction over 16-lane groups) ----
        #pragma unroll
        for (int r = 0; r < 8; ++r) {
            float mx = fmaxf(sv[r], sv[8 + r]);
            #pragma unroll
            for (int off = 8; off >= 1; off >>= 1)
                mx = fmaxf(mx, __shfl_xor(mx, off, 16));
            const float mnew  = fmaxf(mrow[r], mx);
            const float alpha = __expf(mrow[r] - mnew);
            mrow[r] = mnew;
            const float p0 = __expf(sv[r] - mnew);
            const float p1 = __expf(sv[8 + r] - mnew);
            sv[r] = p0; sv[8 + r] = p1;
            float rs = p0 + p1;
            #pragma unroll
            for (int off = 8; off >= 1; off >>= 1)
                rs += __shfl_xor(rs, off, 16);
            lrow[r] = lrow[r] * alpha + rs;
            #pragma unroll
            for (int f = 0; f < 4; ++f) O[f][r] *= alpha;
        }

        // ---- P: C-layout -> A-layout via per-wave LDS slot ----
        #pragma unroll
        for (int t = 0; t < 2; ++t)
            #pragma unroll
            for (int r = 0; r < 8; ++r)
                ((_Float16*)pslot)[(r + 8 * hi) * 32 + t * 16 + nl] = (_Float16)sv[t * 8 + r];
        asm volatile("s_wait_dscnt 0" ::: "memory");

        v16h Pfr;
        {
            const int koff = (lane < 16) ? 0 : 8;
            v8h p0 = *(const v8h*)(pslot + nl * 32 + koff);
            v8h p1 = *(const v8h*)(pslot + nl * 32 + koff + 16);
            Pfr = cat16(p0, p1);
        }

        // ---- O += P * V (V^T rows are contiguous in s) ----
        {
            const int kb = (lane < 16) ? 0 : 8;
            #pragma unroll
            for (int t4 = 0; t4 < 4; ++t4) {
                const _Float16* vp = Vt + ((size_t)((b * 8 + h) * 64 + t4 * 16 + nl) << 10) + j0 + kb;
                v8h b0 = *(const v8h*)(vp);
                v8h b1 = *(const v8h*)(vp + 16);
                O[t4] = wmma_f16(Pfr, cat16(b0, b1), O[t4]);
            }
        }
    }

    // ---- normalize and store X (f16, row-major b,s,d) ----
    #pragma unroll
    for (int r = 0; r < 8; ++r) {
        const float inv = 1.0f / lrow[r];
        const size_t row = (size_t)(b * 1024 + it * 16 + r + 8 * hi) * 512 + h * 64;
        #pragma unroll
        for (int t4 = 0; t4 < 4; ++t4)
            X[row + t4 * 16 + nl] = (_Float16)(O[t4][r] * inv);
    }
}

// ---------------------------------------------------------------------------
extern "C" void kernel_launch(void* const* d_in, const int* in_sizes, int n_in,
                              void* d_out, int out_size, void* d_ws, size_t ws_size,
                              hipStream_t stream) {
    const float* query = (const float*)d_in[0];
    const float* key   = (const float*)d_in[1];
    const float* value = (const float*)d_in[2];
    const float* Wq = (const float*)d_in[3];  const float* bq = (const float*)d_in[4];
    const float* Wk = (const float*)d_in[5];  const float* bk = (const float*)d_in[6];
    const float* Wv = (const float*)d_in[7];  const float* bv = (const float*)d_in[8];
    const float* Wo = (const float*)d_in[9];  const float* bo = (const float*)d_in[10];
    const float* emb = (const float*)d_in[11];
    const int* pc = (const int*)d_in[12];
    const int* sb = (const int*)d_in[13];

    char* ws = (char*)d_ws;
    _Float16* qh  = (_Float16*)(ws);                       //  8 MB  Q (f16, scaled)
    _Float16* kh  = (_Float16*)(ws + (size_t)(8u  << 20)); //  8 MB  K (f16)
    _Float16* vt  = (_Float16*)(ws + (size_t)(16u << 20)); //  8 MB  V^T (f16)
    _Float16* xh  = (_Float16*)(ws + (size_t)(24u << 20)); //  8 MB  attn out (f16)
    int*      cn  = (int*)     (ws + (size_t)(32u << 20)); // 32 MB  packed bias counts
    _Float16* qin = (_Float16*)(ws + (size_t)(64u << 20)); //  8 MB  query (f16)
    _Float16* kin = (_Float16*)(ws + (size_t)(72u << 20)); //  8 MB  key   (f16)
    _Float16* vin = (_Float16*)(ws + (size_t)(80u << 20)); //  8 MB  value (f16)

    hipMemsetAsync(cn, 0, (size_t)32u << 20, stream);
    count_kernel<<<64, 256, 0, stream>>>(pc, sb, cn);

    cvt_kernel<<<2048, 256, 0, stream>>>(query, qin);
    cvt_kernel<<<2048, 256, 0, stream>>>(key,   kin);
    cvt_kernel<<<2048, 256, 0, stream>>>(value, vin);

    dim3 ggrid(64, 8);
    gemm_kernel<0><<<ggrid, 256, 0, stream>>>(qin, Wq, bq, qh);
    gemm_kernel<1><<<ggrid, 256, 0, stream>>>(kin, Wk, bk, kh);
    gemm_kernel<2><<<ggrid, 256, 0, stream>>>(vin, Wv, bv, vt);

    attn_kernel<<<512, 256, 0, stream>>>(qh, kh, vt, cn, emb, xh);

    gemm_kernel<3><<<ggrid, 256, 0, stream>>>(xh, Wo, bo, d_out);
}
